// ChunkedCrossAttention_22600117911814
// MI455X (gfx1250) — compile-verified, matrix operand order checked by reading, not run on hardware
//
#include <hip/hip_runtime.h>
#include <hip/hip_bf16.h>
#include <math.h>

// ---------------------------------------------------------------------------
// Chunked cross attention for MI455X (gfx1250), wave32 + WMMA f16->f32,
// LDS-blocked GEMMs (128x128 block, 64x32 per wave) with async global->LDS.
// B=4 S=2048 D=1024 C=32 N=2 L=128 H=16 DK=64 CHUNK=64
// ---------------------------------------------------------------------------

typedef _Float16 half_t;
typedef __attribute__((ext_vector_type(16))) _Float16 v16h;
typedef __attribute__((ext_vector_type(8)))  _Float16 v8h;
typedef __attribute__((ext_vector_type(8)))  float    v8f;

#define WB 4
#define WS 2048
#define WD 1024
#define WC 32
#define WH 16
#define WDK 64
#define WCHUNK 64
#define WNL 256                       // N*L
#define ROWS_H (WB * WS)              // 8192
#define ROWS_E (WB * WC * WNL)        // 32768
#define VALID_ROWS (WS - WCHUNK + 1)  // 1985
#define WSCALE 0.125f                 // 1/sqrt(64)

#define LDS_STRIDE 40                 // 32 + 8 halves: 16B aligned, bank-spread

#if defined(__AMDGCN__) &&                                              \
    __has_builtin(__builtin_amdgcn_global_load_async_to_lds_b128) &&    \
    __has_builtin(__builtin_amdgcn_s_wait_asynccnt)
#define USE_ASYNC 1
typedef int v4i __attribute__((vector_size(16)));
typedef __attribute__((address_space(1))) v4i as1_v4i;
typedef __attribute__((address_space(3))) v4i as3_v4i;
#else
#define USE_ASYNC 0
#endif

__device__ __forceinline__ v8f wmma16(v16h a, v16h b, v8f c) {
  return __builtin_amdgcn_wmma_f32_16x16x32_f16(false, a, false, b, (short)0, c,
                                                false, false);
}

// Fragment loader for v_wmma_f32_16x16x32_f16. Works for A (MxK row-major,
// lane&15 = M) and B supplied N-major (NxK row-major, lane&15 = N).
// elements 0..7 = K[k0+hi*8 ..], elements 8..15 = K[k0+16+hi*8 ..].
__device__ __forceinline__ v16h load_frag(const half_t* __restrict__ base,
                                          int ld, int k0, int lane) {
  int m  = lane & 15;
  int hi = (lane >> 4) & 1;
  const half_t* p = base + (size_t)m * ld + k0 + hi * 8;
  v8h lo = *(const v8h*)(p);
  v8h hv = *(const v8h*)(p + 16);
  v16h r;
#pragma unroll
  for (int e = 0; e < 8; ++e) { r[e] = lo[e]; r[e + 8] = hv[e]; }
  return r;
}

// Stage one K-step tile (A:128x32, B:128x32 halves) into LDS.
// Each of 256 threads moves 16 halves of A and 16 of B (2x b128 each).
__device__ __forceinline__ void stage_tile(const half_t* __restrict__ gA,
                                           const half_t* __restrict__ gB,
                                           int ldg, half_t* sA, half_t* sB,
                                           int tid) {
  int row = tid >> 1;
  int cs  = (tid & 1) * 16;
  const half_t* pa = gA + (size_t)row * ldg + cs;
  const half_t* pb = gB + (size_t)row * ldg + cs;
  half_t* la = sA + row * LDS_STRIDE + cs;
  half_t* lb = sB + row * LDS_STRIDE + cs;
#if USE_ASYNC
  __builtin_amdgcn_global_load_async_to_lds_b128((as1_v4i*)pa, (as3_v4i*)la, 0,
                                                 0);
  __builtin_amdgcn_global_load_async_to_lds_b128((as1_v4i*)(pa + 8),
                                                 (as3_v4i*)(la + 8), 0, 0);
  __builtin_amdgcn_global_load_async_to_lds_b128((as1_v4i*)pb, (as3_v4i*)lb, 0,
                                                 0);
  __builtin_amdgcn_global_load_async_to_lds_b128((as1_v4i*)(pb + 8),
                                                 (as3_v4i*)(lb + 8), 0, 0);
#else
  v8h a0 = *(const v8h*)pa, a1 = *(const v8h*)(pa + 8);
  v8h b0 = *(const v8h*)pb, b1 = *(const v8h*)(pb + 8);
  *(v8h*)la = a0; *(v8h*)(la + 8) = a1;
  *(v8h*)lb = b0; *(v8h*)(lb + 8) = b1;
#endif
}

__device__ __forceinline__ void async_wait0() {
#if USE_ASYNC
  __builtin_amdgcn_s_wait_asynccnt(0);
#endif
}

// ---------------------------------------------------------------------------
// 1. LayerNorm of h[:,63:] -> f16, padded with zeros to (B, 2048, D)
// ---------------------------------------------------------------------------
__global__ __launch_bounds__(256) void ln_cast_kernel(
    const float* __restrict__ h, const float* __restrict__ gamma,
    const float* __restrict__ beta, half_t* __restrict__ hb16) {
  int row = blockIdx.x;                 // 0..8191
  int b = row >> 11, t = row & 2047;
  half_t* dst = hb16 + (size_t)row * WD;
  int tid = threadIdx.x;
  if (t >= VALID_ROWS) {                // zero padding rows (block-uniform)
    for (int i = tid; i < WD; i += 256) dst[i] = (half_t)0.0f;
    return;
  }
  const float* x = h + ((size_t)b * WS + t + (WCHUNK - 1)) * WD;
  float s = 0.f, s2 = 0.f;
  for (int i = tid; i < WD; i += 256) { float v = x[i]; s += v; s2 += v * v; }
  __shared__ float sh[256], sh2[256];
  sh[tid] = s; sh2[tid] = s2; __syncthreads();
  for (int off = 128; off > 0; off >>= 1) {
    if (tid < off) { sh[tid] += sh[tid + off]; sh2[tid] += sh2[tid + off]; }
    __syncthreads();
  }
  float mu  = sh[0] * (1.0f / WD);
  float var = sh2[0] * (1.0f / WD) - mu * mu;
  float rs  = rsqrtf(var + 1e-5f);
  for (int i = tid; i < WD; i += 256)
    dst[i] = (half_t)((x[i] - mu) * rs * gamma[i] + beta[i]);
}

// ---------------------------------------------------------------------------
// 2. Elementwise f32 -> f16 cast (for e)
// ---------------------------------------------------------------------------
__global__ __launch_bounds__(256) void cast_f16_kernel(
    const float* __restrict__ src, half_t* __restrict__ dst, size_t n) {
  size_t i = (size_t)blockIdx.x * 256 + threadIdx.x;
  size_t stride = (size_t)gridDim.x * 256;
  for (; i < n; i += stride) dst[i] = (half_t)src[i];
}

// ---------------------------------------------------------------------------
// 3. Weight transpose + cast: W (KxN) f32 -> WT (NxK) f16   (1024x1024)
// ---------------------------------------------------------------------------
__global__ __launch_bounds__(256) void transpose_cast_kernel(
    const float* __restrict__ W, half_t* __restrict__ WT) {
  int idx = blockIdx.x * 256 + threadIdx.x;   // coalesced read
  int k = idx >> 10, n = idx & 1023;
  WT[(size_t)n * WD + k] = (half_t)W[idx];
}

// ---------------------------------------------------------------------------
// 4. Blocked WMMA GEMM: block tile 128x128, K-step 32, double-buffered LDS.
//    8 waves, each computing 64x32 (4x2 subtiles, 8 independent accumulators).
//    A (MxK row-major), Bt (NxK row-major).  All dims multiples of 128.
//    MODE 0: C16[row*N+col] = acc + bias[col]                 (q, k proj)
//    MODE 1: vT[(bc,h,dk,j)] = acc + bias[col]                (v proj, T out)
//    MODE 2: out f32 with shift-by-63 + residual              (Wo proj)
// ---------------------------------------------------------------------------
template <int MODE>
__global__ __launch_bounds__(256) void gemm128_kernel(
    const half_t* __restrict__ A, const half_t* __restrict__ Bt,
    const float* __restrict__ bias, half_t* __restrict__ C16,
    const float* __restrict__ hres, float* __restrict__ outf, int M, int N,
    int K) {
  __shared__ __align__(16) half_t sA[2][128 * LDS_STRIDE];
  __shared__ __align__(16) half_t sB[2][128 * LDS_STRIDE];

  int tid  = threadIdx.x;
  int lane = tid & 31, wave = tid >> 5;
  int tm0 = blockIdx.y * 128;
  int tn0 = blockIdx.x * 128;
  int mw = (wave >> 2) * 64;            // wave row offset in block tile
  int nw = (wave & 3) * 32;             // wave col offset in block tile

  const half_t* Ab = A + (size_t)tm0 * K;
  const half_t* Bb = Bt + (size_t)tn0 * K;
  int nk = K >> 5;

  stage_tile(Ab, Bb, K, sA[0], sB[0], tid);
  async_wait0();
  __syncthreads();

  v8f acc[4][2] = {};
  for (int kt = 0; kt < nk; ++kt) {
    int buf = kt & 1;
    if (kt + 1 < nk)
      stage_tile(Ab + (kt + 1) * 32, Bb + (kt + 1) * 32, K, sA[buf ^ 1],
                 sB[buf ^ 1], tid);

    const half_t* As_w = sA[buf] + mw * LDS_STRIDE;
    const half_t* Bs_w = sB[buf] + nw * LDS_STRIDE;
    v16h af[4], bf[2];
#pragma unroll
    for (int i = 0; i < 4; ++i)
      af[i] = load_frag(As_w + i * 16 * LDS_STRIDE, LDS_STRIDE, 0, lane);
#pragma unroll
    for (int j = 0; j < 2; ++j)
      bf[j] = load_frag(Bs_w + j * 16 * LDS_STRIDE, LDS_STRIDE, 0, lane);
#pragma unroll
    for (int i = 0; i < 4; ++i)
#pragma unroll
      for (int j = 0; j < 2; ++j) acc[i][j] = wmma16(af[i], bf[j], acc[i][j]);

    async_wait0();
    __syncthreads();
  }

  // Epilogue: C/D layout -> row = base + (lane>>4)*8 + r, col = base + lane&15
  int nl = lane & 15, mb = (lane >> 4) * 8;
#pragma unroll
  for (int i = 0; i < 4; ++i) {
#pragma unroll
    for (int j = 0; j < 2; ++j) {
      int col = tn0 + nw + j * 16 + nl;
      float bv = bias[col];
#pragma unroll
      for (int r = 0; r < 8; ++r) {
        int row = tm0 + mw + i * 16 + mb + r;
        float v = acc[i][j][r] + bv;
        if (MODE == 0) {
          C16[(size_t)row * N + col] = (half_t)v;
        } else if (MODE == 1) {
          int bc = row >> 8, jj = row & 255;
          int hh = col >> 6, dk = col & 63;
          C16[(((size_t)bc * WH + hh) * WDK + dk) * WNL + jj] = (half_t)v;
        } else {
          int b = row >> 11, t = row & 2047;
          if (t < VALID_ROWS) {          // shift by 63; tail rows discarded
            size_t o = ((size_t)b * WS + t + (WCHUNK - 1)) * WD + col;
            outf[o] = v + hres[o];
          }
        }
      }
    }
  }
}

// ---------------------------------------------------------------------------
// 5. Attention: one block per (b,c,h). Scores (64x256) via WMMA into LDS,
//    softmax over 256, P@V via WMMA (A from LDS, B = vT from global).
// ---------------------------------------------------------------------------
__global__ __launch_bounds__(256) void attention_kernel(
    const half_t* __restrict__ q16, const half_t* __restrict__ k16,
    const half_t* __restrict__ vT16, half_t* __restrict__ att16) {
  int unit = blockIdx.x;                  // 0..2047
  int bc = unit >> 4;                     // b*32 + c
  int hh = unit & 15;
  int tid = threadIdx.x;
  int lane = tid & 31, wave = tid >> 5;

  __shared__ __align__(16) half_t sc[WCHUNK * WNL];   // 64x256 f16 = 32 KB
  __shared__ float rmax[WCHUNK], rinv[WCHUNK];

  const half_t* Aq = q16 + (size_t)bc * WCHUNK * WD + hh * WDK;
  const half_t* Bk = k16 + (size_t)bc * WNL * WD + hh * WDK;

  // --- scores = (q . k^T) * SCALE : 64 tiles (4m x 16n), 8 per wave
#pragma unroll
  for (int it = 0; it < 8; ++it) {
    int tile = wave * 8 + it;
    int tm = tile >> 4, tn = tile & 15;
    const half_t* Ab = Aq + (size_t)tm * 16 * WD;
    const half_t* Bb = Bk + (size_t)tn * 16 * WD;
    v8f acc = {};
    acc = wmma16(load_frag(Ab, WD, 0, lane), load_frag(Bb, WD, 0, lane), acc);
    acc = wmma16(load_frag(Ab, WD, 32, lane), load_frag(Bb, WD, 32, lane), acc);
    int n = lane & 15, mb = (lane >> 4) * 8;
#pragma unroll
    for (int r = 0; r < 8; ++r)
      sc[(tm * 16 + mb + r) * WNL + tn * 16 + n] = (half_t)(acc[r] * WSCALE);
  }
  __syncthreads();

  // --- row stats (thread-per-row)
  if (tid < WCHUNK) {
    float mx = -1e30f;
    for (int j = 0; j < WNL; ++j) mx = fmaxf(mx, (float)sc[tid * WNL + j]);
    float s = 0.f;
    for (int j = 0; j < WNL; ++j) s += __expf((float)sc[tid * WNL + j] - mx);
    rmax[tid] = mx;
    rinv[tid] = 1.0f / s;
  }
  __syncthreads();

  // --- normalize in place (each thread owns its 64 elements)
#pragma unroll 4
  for (int e = 0; e < 64; ++e) {
    int idx = tid * 64 + e;
    int i = idx >> 8;
    sc[idx] = (half_t)(__expf((float)sc[idx] - rmax[i]) * rinv[i]);
  }
  __syncthreads();

  // --- out = P @ V : 16 tiles (4m x 4n over dk), 2 per wave
  const half_t* Bv = vT16 + ((size_t)bc * WH + hh) * WDK * WNL;
#pragma unroll
  for (int it = 0; it < 2; ++it) {
    int tile = wave * 2 + it;
    int tm = tile >> 2, tn = tile & 3;
    const half_t* Ab = sc + (size_t)tm * 16 * WNL;
    const half_t* Bb = Bv + (size_t)tn * 16 * WNL;
    v8f acc = {};
    for (int k0 = 0; k0 < WNL; k0 += 32)
      acc = wmma16(load_frag(Ab, WNL, k0, lane), load_frag(Bb, WNL, k0, lane),
                   acc);
    int n = lane & 15, mb = (lane >> 4) * 8;
#pragma unroll
    for (int r = 0; r < 8; ++r)
      att16[((size_t)bc * WCHUNK + tm * 16 + mb + r) * WD + hh * WDK +
            tn * 16 + n] = (half_t)acc[r];
  }
}

// ---------------------------------------------------------------------------
// 6. Residual-only head rows: out[b,s,:] = h[b,s,:] for s < 63
// ---------------------------------------------------------------------------
__global__ __launch_bounds__(256) void residual_head_kernel(
    const float* __restrict__ h, float* __restrict__ out) {
  int idx = blockIdx.x * 256 + threadIdx.x;     // 4*63*1024 = 258048 exact
  int b = idx / ((WCHUNK - 1) * WD);
  int rem = idx % ((WCHUNK - 1) * WD);
  int s = rem / WD, d = rem % WD;
  size_t o = ((size_t)b * WS + s) * WD + d;
  out[o] = h[o];
}

// ---------------------------------------------------------------------------
// Host launcher
// ---------------------------------------------------------------------------
extern "C" void kernel_launch(void* const* d_in, const int* in_sizes, int n_in,
                              void* d_out, int out_size, void* d_ws,
                              size_t ws_size, hipStream_t stream) {
  const float* h     = (const float*)d_in[0];
  const float* e     = (const float*)d_in[1];
  const float* Wq    = (const float*)d_in[2];
  const float* bq    = (const float*)d_in[3];
  const float* Wk    = (const float*)d_in[4];
  const float* bk    = (const float*)d_in[5];
  const float* Wv    = (const float*)d_in[6];
  const float* bv    = (const float*)d_in[7];
  const float* Wo    = (const float*)d_in[8];
  const float* bo    = (const float*)d_in[9];
  const float* gamma = (const float*)d_in[10];
  const float* beta  = (const float*)d_in[11];
  float* out = (float*)d_out;

  // Workspace layout (f16 elements), ~248 MB total
  const size_t HB  = (size_t)ROWS_H * WD;   //  8,388,608
  const size_t E16 = (size_t)ROWS_E * WD;   // 33,554,432
  const size_t WSZ = (size_t)WD * WD;       //  1,048,576
  half_t* w = (half_t*)d_ws;
  half_t* hb16  = w;                 w += HB;
  half_t* e16   = w;                 w += E16;
  half_t* WqT   = w;                 w += WSZ;
  half_t* WkT   = w;                 w += WSZ;
  half_t* WvT   = w;                 w += WSZ;
  half_t* WoT   = w;                 w += WSZ;
  half_t* q16   = w;                 w += HB;
  half_t* k16   = w;                 w += E16;
  half_t* vT16  = w;                 w += E16;
  half_t* att16 = w;                 w += HB;

  // 1. LayerNorm + pad + cast
  ln_cast_kernel<<<ROWS_H, 256, 0, stream>>>(h, gamma, beta, hb16);
  // 2. cast e
  cast_f16_kernel<<<8192, 256, 0, stream>>>(e, e16, E16);
  // 3. transpose-cast weights
  transpose_cast_kernel<<<4096, 256, 0, stream>>>(Wq, WqT);
  transpose_cast_kernel<<<4096, 256, 0, stream>>>(Wk, WkT);
  transpose_cast_kernel<<<4096, 256, 0, stream>>>(Wv, WvT);
  transpose_cast_kernel<<<4096, 256, 0, stream>>>(Wo, WoT);
  // 4. q projection: (8192x1024) @ (1024x1024)
  gemm128_kernel<0><<<dim3(WD / 128, ROWS_H / 128), 256, 0, stream>>>(
      hb16, WqT, bq, q16, nullptr, nullptr, ROWS_H, WD, WD);
  // 5. k projection: (32768x1024) @ (1024x1024)
  gemm128_kernel<0><<<dim3(WD / 128, ROWS_E / 128), 256, 0, stream>>>(
      e16, WkT, bk, k16, nullptr, nullptr, ROWS_E, WD, WD);
  // 6. v projection, transposed output
  gemm128_kernel<1><<<dim3(WD / 128, ROWS_E / 128), 256, 0, stream>>>(
      e16, WvT, bv, vT16, nullptr, nullptr, ROWS_E, WD, WD);
  // 7. attention per (b,c,h)
  attention_kernel<<<WB * WC * WH, 256, 0, stream>>>(q16, k16, vT16, att16);
  // 8. residual-only head rows
  residual_head_kernel<<<(WB * (WCHUNK - 1) * WD) / 256, 256, 0, stream>>>(h,
                                                                           out);
  // 9. output projection + shift + residual, fused into d_out
  gemm128_kernel<2><<<dim3(WD / 128, ROWS_H / 128), 256, 0, stream>>>(
      att16, WoT, bo, nullptr, h, out, ROWS_H, WD, WD);
}
